// MetaMoE_28089086116694
// MI455X (gfx1250) — matrix operand order repaired and checked by reading, not compiled, not run on hardware
//
#include <hip/hip_runtime.h>
#include <math.h>

#define BDIM 8192
#define INDIM 1024
#define HIDD 2048
#define NE 16
#define GDIM 256

typedef __attribute__((ext_vector_type(16))) __bf16 v16bf;
typedef __attribute__((ext_vector_type(8)))  __bf16 v8bf;
typedef __attribute__((ext_vector_type(8)))  float  v8f;

// ---------------- LayerNorm: x -> xhat(bf16), xg = xhat*g_ln_w+g_ln_b (bf16)
__global__ void __launch_bounds__(256)
ln_kernel(const float* __restrict__ x, const float* __restrict__ gw,
          const float* __restrict__ gb, __bf16* __restrict__ xhat,
          __bf16* __restrict__ xg) {
  __shared__ float s1[256], s2[256];
  int row = blockIdx.x;
  int t = threadIdx.x;
  const float* xr = x + (size_t)row * INDIM;
  float vals[4];
  float v0 = 0.f, v1 = 0.f;
#pragma unroll
  for (int i = 0; i < 4; i++) {
    float v = xr[t + 256 * i];
    vals[i] = v; v0 += v; v1 += v * v;
  }
  s1[t] = v0; s2[t] = v1;
  __syncthreads();
  for (int off = 128; off > 0; off >>= 1) {
    if (t < off) { s1[t] += s1[t + off]; s2[t] += s2[t + off]; }
    __syncthreads();
  }
  float mu = s1[0] * (1.0f / INDIM);
  float var = s2[0] * (1.0f / INDIM) - mu * mu;
  float rs = rsqrtf(var + 1e-5f);
#pragma unroll
  for (int i = 0; i < 4; i++) {
    int c = t + 256 * i;
    float xh = (vals[i] - mu) * rs;
    xhat[(size_t)row * INDIM + c] = (__bf16)xh;
    xg[(size_t)row * INDIM + c] = (__bf16)(xh * gw[c] + gb[c]);
  }
}

// ---------------- Repack f32 KxN weight into WMMA-B fragment order (bf16)
// tile (kt,nt): lane = ((kr/8)%2)*16 + (n%16), 16 contiguous bf16 per lane, p = kr%8 + 8*(kr/16)
__global__ void __launch_bounds__(256)
pack_b_kernel(const float* __restrict__ W, __bf16* __restrict__ out, int K, int N) {
  size_t idx = (size_t)blockIdx.x * 256 + threadIdx.x;
  if (idx >= (size_t)K * N) return;
  int n = (int)(idx % N);
  int k = (int)(idx / N);
  int kt = k >> 5, kr = k & 31;
  int nt = n >> 4, nl = n & 15;
  int lane = (((kr >> 3) & 1) << 4) | nl;
  int p = (kr & 7) | ((kr >> 4) << 3);
  int nTiles = N >> 4;
  size_t o = (((size_t)kt * nTiles + nt) * 32 + lane) * 16 + p;
  out[o] = (__bf16)W[idx];
}

// ---------------- Per-expert A affine: xe = xhat * e_ln_w[e] + e_ln_b[e] (bf16)
__global__ void __launch_bounds__(256)
xe_kernel(const __bf16* __restrict__ xhat, const float* __restrict__ lnw,
          const float* __restrict__ lnb, __bf16* __restrict__ xe) {
  size_t idx = (size_t)blockIdx.x * 256 + threadIdx.x;
  int c = (int)(idx & (INDIM - 1));
  float v = (float)xhat[idx];
  xe[idx] = (__bf16)(v * lnw[c] + lnb[c]);
}

// ---------------- Generic bf16 WMMA GEMM: C = act(A @ B + bias), bf16 out
// wave tile: 32 rows x 64 cols (2 M-frags x 4 N-frags = 8 WMMA per K-step).
// Grid: x = N/64, y = M/256, block = 256 (8 waves).
__global__ void __launch_bounds__(256)
gemm_bf16_wmma(const __bf16* __restrict__ A, const __bf16* __restrict__ Bp,
               const float* __restrict__ bias, __bf16* __restrict__ C,
               int N, int K, int relu) {
  int lane = threadIdx.x & 31;
  int wave = threadIdx.x >> 5;
  int rowBase = (blockIdx.y * 8 + wave) * 32;
  int ntBase = blockIdx.x * 4;
  int nTiles = N >> 4;
  int r0 = rowBase + (lane & 15);
  int khalf = (lane >> 4) * 8;  // A 16-bit layout: K-halves {0..7,16..23} / {8..15,24..31}
  const __bf16* arow0 = A + (size_t)r0 * K + khalf;
  const __bf16* arow1 = arow0 + (size_t)16 * K;
  v8f acc[2][4] = {};
  int KT = K >> 5;
  for (int kt = 0; kt < KT; kt++) {
    __builtin_prefetch(arow0 + 32, 0, 3);
    __builtin_prefetch(arow1 + 32, 0, 3);
    v8bf a00 = *(const v8bf*)(arow0);
    v8bf a01 = *(const v8bf*)(arow0 + 16);
    v8bf a10 = *(const v8bf*)(arow1);
    v8bf a11 = *(const v8bf*)(arow1 + 16);
    arow0 += 32; arow1 += 32;
    v16bf af0, af1;
#pragma unroll
    for (int i = 0; i < 8; i++) {
      af0[i] = a00[i]; af0[i + 8] = a01[i];
      af1[i] = a10[i]; af1[i + 8] = a11[i];
    }
    const __bf16* bbase = Bp + (((size_t)kt * nTiles + ntBase) * 32 + lane) * 16;
#pragma unroll
    for (int nt = 0; nt < 4; nt++) {
      v16bf b = *(const v16bf*)(bbase + (size_t)nt * 512);
      acc[0][nt] = __builtin_amdgcn_wmma_f32_16x16x32_bf16(false, af0, false, b, (short)0, acc[0][nt], false, false);
      acc[1][nt] = __builtin_amdgcn_wmma_f32_16x16x32_bf16(false, af1, false, b, (short)0, acc[1][nt], false, false);
    }
  }
  // D layout: lanes 0-15 -> N=lane, M=vgpr; lanes 16-31 -> N=lane-16, M=vgpr+8
  int col0 = (ntBase << 4) + (lane & 15);
  int rowOff = (lane >> 4) * 8;
#pragma unroll
  for (int mi = 0; mi < 2; mi++) {
    int rb = rowBase + mi * 16 + rowOff;
#pragma unroll
    for (int nt = 0; nt < 4; nt++) {
      int col = col0 + nt * 16;
      float bv = bias[col];
#pragma unroll
      for (int rr = 0; rr < 8; rr++) {
        float v = acc[mi][nt][rr] + bv;
        if (relu) v = v > 0.f ? v : 0.f;
        C[(size_t)(rb + rr) * N + col] = (__bf16)v;
      }
    }
  }
}

// ---------------- Gate GEMM2 (K=256, N=16) + row softmax over 16 experts
__global__ void __launch_bounds__(256)
gate2_softmax(const __bf16* __restrict__ gh, const __bf16* __restrict__ Bp,
              const float* __restrict__ b2, float* __restrict__ gate_w) {
  int lane = threadIdx.x & 31;
  int wave = threadIdx.x >> 5;
  int rowBase = (blockIdx.x * 8 + wave) * 16;
  int r = rowBase + (lane & 15);
  int khalf = (lane >> 4) * 8;
  const __bf16* arow = gh + (size_t)r * GDIM + khalf;
  v8f acc = {};
  for (int kt = 0; kt < GDIM / 32; kt++) {
    v8bf a0 = *(const v8bf*)(arow);
    v8bf a1 = *(const v8bf*)(arow + 16);
    arow += 32;
    v16bf af;
#pragma unroll
    for (int i = 0; i < 8; i++) { af[i] = a0[i]; af[i + 8] = a1[i]; }
    v16bf bfr = *(const v16bf*)(Bp + ((size_t)kt * 32 + lane) * 16);
    acc = __builtin_amdgcn_wmma_f32_16x16x32_bf16(false, af, false, bfr, (short)0, acc, false, false);
  }
  int n = lane & 15;
  int rowOff = (lane >> 4) * 8;
  float bb = b2[n];
#pragma unroll
  for (int rr = 0; rr < 8; rr++) {
    float v = acc[rr] + bb;
    float m = v;
    for (int s = 1; s < 16; s <<= 1) m = fmaxf(m, __shfl_xor(m, s, 16));
    float e = __expf(v - m);
    float sum = e;
    for (int s = 1; s < 16; s <<= 1) sum += __shfl_xor(sum, s, 16);
    gate_w[(size_t)(rowBase + rowOff + rr) * NE + n] = e / sum;
  }
}

// ---------------- Expert head (256->2) fused with gated accumulation
__global__ void __launch_bounds__(256)
accum_kernel(const __bf16* __restrict__ h2, const float* __restrict__ w3,
             const float* __restrict__ b3, const float* __restrict__ gate_w,
             float* __restrict__ acc, int e) {
  int b = blockIdx.x * 256 + threadIdx.x;
  const __bf16* hr = h2 + (size_t)b * GDIM;
  float s0 = 0.f, s1 = 0.f;
#pragma unroll 4
  for (int g = 0; g < GDIM; g++) {
    float h = (float)hr[g];
    s0 += h * w3[g * 2 + 0];
    s1 += h * w3[g * 2 + 1];
  }
  float gw = gate_w[(size_t)b * NE + e];
  acc[b * 2 + 0] += gw * (s0 + b3[0]);
  acc[b * 2 + 1] += gw * (s1 + b3[1]);
}

__global__ void __launch_bounds__(256) init_acc(float* acc) {
  int i = blockIdx.x * 256 + threadIdx.x;
  acc[i] = 0.f;
}

__global__ void __launch_bounds__(256)
final_kernel(const float* __restrict__ acc, float* __restrict__ out) {
  int b = blockIdx.x * 256 + threadIdx.x;
  out[b] = acc[b * 2 + 0];
  float z = acc[b * 2 + 1];
  float sp = (z > 20.f) ? z : log1pf(__expf(z));
  out[BDIM + b] = sp + 1e-6f;
}

extern "C" void kernel_launch(void* const* d_in, const int* in_sizes, int n_in,
                              void* d_out, int out_size, void* d_ws, size_t ws_size,
                              hipStream_t stream) {
  const float* x    = (const float*)d_in[0];
  const float* glnw = (const float*)d_in[1];
  const float* glnb = (const float*)d_in[2];
  const float* gw1  = (const float*)d_in[3];
  const float* gb1  = (const float*)d_in[4];
  const float* gw2  = (const float*)d_in[5];
  const float* gb2  = (const float*)d_in[6];
  const float* elnw = (const float*)d_in[7];
  const float* elnb = (const float*)d_in[8];
  const float* ew1  = (const float*)d_in[9];
  const float* eb1  = (const float*)d_in[10];
  const float* ew2  = (const float*)d_in[11];
  const float* eb2  = (const float*)d_in[12];
  const float* ew3  = (const float*)d_in[13];
  const float* eb3  = (const float*)d_in[14];
  float* out = (float*)d_out;

  char* ws = (char*)d_ws;
  size_t off = 0;
  auto alloc = [&](size_t bytes) -> char* {
    char* p = ws + off;
    off += (bytes + 255) & ~(size_t)255;
    return p;
  };
  __bf16* xhat = (__bf16*)alloc((size_t)BDIM * INDIM * 2);
  __bf16* xg   = (__bf16*)alloc((size_t)BDIM * INDIM * 2);
  __bf16* gw1p = (__bf16*)alloc((size_t)INDIM * GDIM * 2);
  __bf16* gh   = (__bf16*)alloc((size_t)BDIM * GDIM * 2);
  __bf16* gw2p = (__bf16*)alloc((size_t)GDIM * NE * 2);
  float*  gatew= (float*) alloc((size_t)BDIM * NE * 4);
  __bf16* xe   = (__bf16*)alloc((size_t)BDIM * INDIM * 2);
  __bf16* w1p  = (__bf16*)alloc((size_t)INDIM * HIDD * 2);
  __bf16* h1   = (__bf16*)alloc((size_t)BDIM * HIDD * 2);
  __bf16* w2p  = (__bf16*)alloc((size_t)HIDD * GDIM * 2);
  __bf16* h2   = (__bf16*)alloc((size_t)BDIM * GDIM * 2);
  float*  acc  = (float*) alloc((size_t)BDIM * 2 * 4);

  // LayerNorm + gate-scaled activations
  ln_kernel<<<BDIM, 256, 0, stream>>>(x, glnw, glnb, xhat, xg);

  // Gate path: gh = relu(xg @ g_w1 + g_b1); gate_w = softmax(gh @ g_w2 + g_b2)
  pack_b_kernel<<<(INDIM * GDIM + 255) / 256, 256, 0, stream>>>(gw1, gw1p, INDIM, GDIM);
  gemm_bf16_wmma<<<dim3(GDIM / 64, BDIM / 256), 256, 0, stream>>>(xg, gw1p, gb1, gh, GDIM, INDIM, 1);
  pack_b_kernel<<<(GDIM * NE + 255) / 256, 256, 0, stream>>>(gw2, gw2p, GDIM, NE);
  gate2_softmax<<<BDIM / 128, 256, 0, stream>>>(gh, gw2p, gb2, gatew);

  init_acc<<<(BDIM * 2) / 256, 256, 0, stream>>>(acc);

  // Expert loop: h1 = relu(xe@W1+b1); h2 = relu(h1@W2+b2); acc += gate*(h2@W3+b3)
  for (int e = 0; e < NE; e++) {
    xe_kernel<<<(BDIM * INDIM) / 256, 256, 0, stream>>>(
        xhat, elnw + (size_t)e * INDIM, elnb + (size_t)e * INDIM, xe);
    pack_b_kernel<<<((size_t)INDIM * HIDD + 255) / 256, 256, 0, stream>>>(
        ew1 + (size_t)e * INDIM * HIDD, w1p, INDIM, HIDD);
    gemm_bf16_wmma<<<dim3(HIDD / 64, BDIM / 256), 256, 0, stream>>>(
        xe, w1p, eb1 + (size_t)e * HIDD, h1, HIDD, INDIM, 1);
    pack_b_kernel<<<((size_t)HIDD * GDIM + 255) / 256, 256, 0, stream>>>(
        ew2 + (size_t)e * HIDD * GDIM, w2p, HIDD, GDIM);
    gemm_bf16_wmma<<<dim3(GDIM / 64, BDIM / 256), 256, 0, stream>>>(
        h1, w2p, eb2 + (size_t)e * GDIM, h2, GDIM, HIDD, 1);
    accum_kernel<<<BDIM / 256, 256, 0, stream>>>(
        h2, ew3 + (size_t)e * GDIM * 2, eb3 + (size_t)e * 2, gatew, acc, e);
  }

  final_kernel<<<BDIM / 256, 256, 0, stream>>>(acc, out);
}
